// SGIN_71425306133015
// MI455X (gfx1250) — matrix operand order repaired and checked by reading, not compiled, not run on hardware
//
#include <hip/hip_runtime.h>

#define N_NODES 50000
#define N_EDGES 800000
#define DCH 64
#define NLAYERS 3
#define NGRAPH 512
#define SCH 32
#define BN_EPS 1e-5f
#define PACK_ELEMS (NLAYERS * 4 * 2 * 32 * 16) /* 12288 bf16 per weight matrix */

typedef __attribute__((ext_vector_type(16))) __bf16    v16bf;
typedef __attribute__((ext_vector_type(2)))  __bf16    v2bf;
typedef __attribute__((ext_vector_type(8)))  float     v8f;
typedef __attribute__((ext_vector_type(8)))  unsigned  v8u;
typedef __attribute__((ext_vector_type(4)))  unsigned  v4u;

// Native fptrunc f32->bf16: let the gfx1250 backend pick its hardware convert.
__device__ __forceinline__ unsigned short f2bfu(float f) {
    __bf16 b = (__bf16)f;
    return __builtin_bit_cast(unsigned short, b);
}

// Pack two floats into one dword of bf16 pair (element order: lo = even element).
__device__ __forceinline__ unsigned pk_bf16(float lo, float hi) {
    v2bf p;
    p[0] = (__bf16)lo;
    p[1] = (__bf16)hi;
    return __builtin_bit_cast(unsigned, p);
}

__device__ __forceinline__ v16bf dw_to_v16bf(v8u w) {
    return __builtin_bit_cast(v16bf, w);
}

__device__ __forceinline__ v8f wmma_bf16(v16bf a, v16bf b, v8f c) {
    // (neg_a, A, neg_b, B, c_mod, C, reuse_a, reuse_b)
    return __builtin_amdgcn_wmma_f32_16x16x32_bf16(false, a, false, b, (short)0, c,
                                                   false, false);
}

// ---------------- prep: pack weights into WMMA B layout, fold BN ----------------
// B-operand layout (bf16 32x16 per K-half): koff = kh*32 (+8 for lanes 16-31),
// e0..7 -> K = koff+t, e8..15 -> K = koff+16+(t-8); N = nb*16 + (lane&15).
__global__ void prep_kernel(const float* __restrict__ W1, const float* __restrict__ W2,
                            const float* __restrict__ b1, const float* __restrict__ gamma,
                            const float* __restrict__ beta, const float* __restrict__ mean,
                            const float* __restrict__ var, unsigned short* __restrict__ w1p,
                            unsigned short* __restrict__ w2p, float* __restrict__ snA,
                            float* __restrict__ snB) {
    int tid = blockIdx.x * blockDim.x + threadIdx.x;
    if (tid < 2 * PACK_ELEMS) {
        const float* W = (tid < PACK_ELEMS) ? W1 : W2;
        unsigned short* wp = (tid < PACK_ELEMS) ? w1p : w2p;
        int t0    = tid % PACK_ELEMS;
        int t     = t0 & 15;
        int lane  = (t0 >> 4) & 31;
        int kh    = (t0 >> 9) & 1;
        int nb    = (t0 >> 10) & 3;
        int layer = t0 >> 12;
        int K  = kh * 32 + ((lane >= 16) ? 8 : 0) + ((t < 8) ? t : t + 8);
        int Nc = nb * 16 + (lane & 15);
        wp[t0] = f2bfu(W[layer * DCH * DCH + K * DCH + Nc]);
    } else if (tid < 2 * PACK_ELEMS + NLAYERS * DCH) {
        int t0 = tid - 2 * PACK_ELEMS;
        int n = t0 % DCH, i = t0 / DCH;
        float a = gamma[i * DCH + n] * rsqrtf(var[i * DCH + n] + BN_EPS);
        snA[t0] = a;
        snB[t0] = a * (b1[i * DCH + n] - mean[i * DCH + n]) + beta[i * DCH + n];
    }
}

__global__ void zero_kernel(float* __restrict__ p, int n4) {
    int i = blockIdx.x * blockDim.x + threadIdx.x;
    if (i < n4) ((float4*)p)[i] = make_float4(0.f, 0.f, 0.f, 0.f);
}

// ---------------- edge aggregation: agg[dst] += w * h[src] ----------------
__global__ void agg_kernel(const float* __restrict__ h, const int* __restrict__ ei,
                           const float* __restrict__ ew, float* __restrict__ agg) {
    int tid = blockIdx.x * blockDim.x + threadIdx.x;
    if (tid >= N_EDGES * 16) return;
    int e = tid >> 4, c = (tid & 15) << 2;
    int s = ei[e], d = ei[N_EDGES + e];
    float w  = ew[e];
    float4 v = *(const float4*)(h + (size_t)s * DCH + c);
    float* ap = agg + (size_t)d * DCH + c;
    atomicAdd(ap + 0, v.x * w);
    atomicAdd(ap + 1, v.y * w);
    atomicAdd(ap + 2, v.z * w);
    atomicAdd(ap + 3, v.w * w);
}

// ---------------- fused GIN node update: z=(1+eps)h+agg; MLP via WMMA ----------------
// One wave32 per 16-row tile. A-operand layout (bf16 16x32): lane<16 -> M=lane,
// e0..7 = K koff..koff+7, e8..15 = K koff+16..koff+23 with koff = kh*32 (+8 for lanes>=16).
// C/D layout: VGPR r -> (M = r + 8*(lane>=16), N = lane&15) within an N-block.
__global__ void __launch_bounds__(32) layer_kernel(
    const float* __restrict__ hsrc, const float* __restrict__ agg,
    const unsigned short* __restrict__ w1p, const unsigned short* __restrict__ w2p,
    const float* __restrict__ snA, const float* __restrict__ snB,
    const float* __restrict__ b2, const float* __restrict__ epsv,
    float* __restrict__ hdst, int layer) {
    __shared__ unsigned short yt[16 * DCH];  // 2 KB intermediate tile (bf16)
    const int lane = threadIdx.x;
    const int tile = blockIdx.x;
    const int row  = lane & 15;
    const int up   = (lane >> 4) & 1;
    const int rowg = tile * 16 + row;
    const float e1 = 1.0f + epsv[layer];

    const float* hrow = hsrc + (size_t)rowg * DCH;
    const float* grow = agg + (size_t)rowg * DCH;

    v8f acc[4];
#pragma unroll
    for (int nb = 0; nb < 4; nb++) acc[nb] = (v8f){0.f, 0.f, 0.f, 0.f, 0.f, 0.f, 0.f, 0.f};

#pragma unroll
    for (int kh = 0; kh < 2; kh++) {
        const int koff = kh * 32 + up * 8;
        float hv[16], gv[16];
        *(float4*)&hv[0]  = *(const float4*)(hrow + koff);
        *(float4*)&hv[4]  = *(const float4*)(hrow + koff + 4);
        *(float4*)&hv[8]  = *(const float4*)(hrow + koff + 16);
        *(float4*)&hv[12] = *(const float4*)(hrow + koff + 20);
        *(float4*)&gv[0]  = *(const float4*)(grow + koff);
        *(float4*)&gv[4]  = *(const float4*)(grow + koff + 4);
        *(float4*)&gv[8]  = *(const float4*)(grow + koff + 16);
        *(float4*)&gv[12] = *(const float4*)(grow + koff + 20);
        v8u aw;
#pragma unroll
        for (int j = 0; j < 8; j++)
            aw[j] = pk_bf16(e1 * hv[2 * j] + gv[2 * j], e1 * hv[2 * j + 1] + gv[2 * j + 1]);
        v16bf A = dw_to_v16bf(aw);
#pragma unroll
        for (int nb = 0; nb < 4; nb++) {
            v16bf B = dw_to_v16bf(
                *(const v8u*)(w1p + (size_t)((((layer * 4 + nb) * 2 + kh) * 32 + lane) * 16)));
            acc[nb] = wmma_bf16(A, B, acc[nb]);
        }
    }
    // fused BN (scale/shift) + ReLU, stage y tile into LDS as bf16 row-major
#pragma unroll
    for (int nb = 0; nb < 4; nb++) {
        const int Nc = nb * 16 + row;
        const float sa = snA[layer * DCH + Nc];
        const float sb = snB[layer * DCH + Nc];
#pragma unroll
        for (int r = 0; r < 8; r++) {
            const int M = r + up * 8;
            float y = sa * acc[nb][r] + sb;
            yt[M * DCH + Nc] = f2bfu(y > 0.f ? y : 0.f);
        }
    }
    __syncthreads();

    v8f acc2[4];
#pragma unroll
    for (int nb = 0; nb < 4; nb++) acc2[nb] = (v8f){0.f, 0.f, 0.f, 0.f, 0.f, 0.f, 0.f, 0.f};
#pragma unroll
    for (int kh = 0; kh < 2; kh++) {
        const int koff = kh * 32 + up * 8;
        // two 16B-aligned ds_load_b128 cover the lane's 16 A elements for this K-half
        v4u ya = *(const v4u*)&yt[row * DCH + koff];
        v4u yb = *(const v4u*)&yt[row * DCH + koff + 16];
        v8u aw;
        aw[0] = ya[0]; aw[1] = ya[1]; aw[2] = ya[2]; aw[3] = ya[3];
        aw[4] = yb[0]; aw[5] = yb[1]; aw[6] = yb[2]; aw[7] = yb[3];
        v16bf A = dw_to_v16bf(aw);
#pragma unroll
        for (int nb = 0; nb < 4; nb++) {
            v16bf B = dw_to_v16bf(
                *(const v8u*)(w2p + (size_t)((((layer * 4 + nb) * 2 + kh) * 32 + lane) * 16)));
            acc2[nb] = wmma_bf16(A, B, acc2[nb]);
        }
    }
    // bias + ReLU + store next h (in-place per-tile safe)
#pragma unroll
    for (int nb = 0; nb < 4; nb++) {
        const int Nc = nb * 16 + row;
        const float bb = b2[layer * DCH + Nc];
#pragma unroll
        for (int r = 0; r < 8; r++) {
            const int M = r + up * 8;
            float y = acc2[nb][r] + bb;
            hdst[(size_t)(tile * 16 + M) * DCH + Nc] = y > 0.f ? y : 0.f;
        }
    }
}

// ---------------- pooling ----------------
__global__ void pool_add_kernel(const float* __restrict__ h, const int* __restrict__ batch,
                                const float* __restrict__ mask, float* __restrict__ hcat,
                                int layer) {
    int tid = blockIdx.x * blockDim.x + threadIdx.x;
    if (tid >= N_NODES * 16) return;
    int n = tid >> 4, c = (tid & 15) << 2;
    int g   = batch[n];
    float m = mask[n];
    float4 v = *(const float4*)(h + (size_t)n * DCH + c);
    float* p = hcat + (size_t)g * (2 * DCH * NLAYERS) + layer * 2 * DCH + c;
    atomicAdd(p + 0, v.x * m);
    atomicAdd(p + 1, v.y * m);
    atomicAdd(p + 2, v.z * m);
    atomicAdd(p + 3, v.w * m);
}

__global__ void pool_center_kernel(const float* __restrict__ h, const int* __restrict__ mapping,
                                   float* __restrict__ hcat, int layer) {
    int tid = blockIdx.x * blockDim.x + threadIdx.x;
    if (tid >= NGRAPH * 16) return;
    int g = tid >> 4, c = (tid & 15) << 2;
    int n = mapping[g];
    float4 v = *(const float4*)(h + (size_t)n * DCH + c);
    *(float4*)(hcat + (size_t)g * (2 * DCH * NLAYERS) + layer * 2 * DCH + DCH + c) = v;
}

// ---------------- final linear [G,384] @ [384,32] + b ----------------
__global__ void final_kernel(const float* __restrict__ hcat, const float* __restrict__ lw,
                             const float* __restrict__ lb, float* __restrict__ out) {
    int tid = blockIdx.x * blockDim.x + threadIdx.x;
    if (tid >= NGRAPH * SCH) return;
    int g = tid / SCH, s = tid % SCH;
    const float* hr = hcat + (size_t)g * (2 * DCH * NLAYERS);
    float acc = lb[s];
    for (int k = 0; k < 2 * DCH * NLAYERS; k++) acc += hr[k] * lw[k * SCH + s];
    out[tid] = acc;
}

extern "C" void kernel_launch(void* const* d_in, const int* in_sizes, int n_in,
                              void* d_out, int out_size, void* d_ws, size_t ws_size,
                              hipStream_t stream) {
    const float* x     = (const float*)d_in[0];
    const float* ew    = (const float*)d_in[1];
    const float* mask  = (const float*)d_in[2];
    const float* W1    = (const float*)d_in[3];
    const float* b1    = (const float*)d_in[4];
    const float* gamma = (const float*)d_in[5];
    const float* beta  = (const float*)d_in[6];
    const float* mean  = (const float*)d_in[7];
    const float* var   = (const float*)d_in[8];
    const float* W2    = (const float*)d_in[9];
    const float* b2    = (const float*)d_in[10];
    const float* epsv  = (const float*)d_in[11];
    const float* lw    = (const float*)d_in[12];
    const float* lb    = (const float*)d_in[13];
    const int* ei      = (const int*)d_in[14];
    const int* batch   = (const int*)d_in[15];
    const int* mapping = (const int*)d_in[16];
    float* outp = (float*)d_out;

    char* ws = (char*)d_ws;
    size_t off = 0;
    float* agg  = (float*)(ws + off); off += (size_t)N_NODES * DCH * 4;
    float* hbuf = (float*)(ws + off); off += (size_t)N_NODES * DCH * 4;
    float* hcat = (float*)(ws + off); off += (size_t)NGRAPH * 2 * DCH * NLAYERS * 4;
    unsigned short* w1p = (unsigned short*)(ws + off); off += (size_t)PACK_ELEMS * 2;
    unsigned short* w2p = (unsigned short*)(ws + off); off += (size_t)PACK_ELEMS * 2;
    float* snA = (float*)(ws + off); off += (size_t)NLAYERS * DCH * 4;
    float* snB = (float*)(ws + off); off += (size_t)NLAYERS * DCH * 4;

    prep_kernel<<<(2 * PACK_ELEMS + NLAYERS * DCH + 255) / 256, 256, 0, stream>>>(
        W1, W2, b1, gamma, beta, mean, var, w1p, w2p, snA, snB);
    zero_kernel<<<((NGRAPH * 2 * DCH * NLAYERS / 4) + 255) / 256, 256, 0, stream>>>(
        hcat, NGRAPH * 2 * DCH * NLAYERS / 4);

    for (int i = 0; i < NLAYERS; i++) {
        const float* hsrc = (i == 0) ? x : hbuf;
        zero_kernel<<<((N_NODES * DCH / 4) + 255) / 256, 256, 0, stream>>>(agg,
                                                                           N_NODES * DCH / 4);
        agg_kernel<<<(N_EDGES * 16 + 255) / 256, 256, 0, stream>>>(hsrc, ei, ew, agg);
        layer_kernel<<<N_NODES / 16, 32, 0, stream>>>(hsrc, agg, w1p, w2p, snA, snB, b2, epsv,
                                                      hbuf, i);
        pool_add_kernel<<<(N_NODES * 16 + 255) / 256, 256, 0, stream>>>(hbuf, batch, mask, hcat,
                                                                        i);
        pool_center_kernel<<<(NGRAPH * 16 + 255) / 256, 256, 0, stream>>>(hbuf, mapping, hcat,
                                                                          i);
    }
    final_kernel<<<(NGRAPH * SCH + 255) / 256, 256, 0, stream>>>(hcat, lw, lb, outp);
}
